// PolicyGradient_28819230556839
// MI455X (gfx1250) — compile-verified, matrix operand order checked by reading, not run on hardware
//
#include <hip/hip_runtime.h>
#include <hip/hip_bf16.h>
#include <math.h>

// CDNA5 WMMA vector types (probe-confirmed signatures)
typedef __attribute__((ext_vector_type(16))) _Float16 v16h;
typedef __attribute__((ext_vector_type(8)))  float    v8f;

#define BATCH   8192
#define S       9
#define D       3
#define H       3
#define DFF     2048
#define NLAYER  3
#define LN_EPS  1e-5f
#define BT      16          // batch items per wave
#define TOK     (BT * S)    // 144 tokens per wave
#define MTILES  (TOK / 16)  // 9 WMMA M-tiles
#define NCHUNK  (DFF / 16)  // 128 WMMA N-chunks over DFF

__launch_bounds__(32)
__global__ void PolicyGradient_kernel(
    const float* __restrict__ x,      // [B][27]
    const float* __restrict__ Wqkv,   // [L][9][3]
    const float* __restrict__ bqkv,   // [L][9]
    const float* __restrict__ Wo,     // [L][3][3]
    const float* __restrict__ bo,     // [L][3]
    const float* __restrict__ g1,     // [L][3]
    const float* __restrict__ beta1,  // [L][3]
    const float* __restrict__ W1,     // [L][2048][3]
    const float* __restrict__ bf1,    // [L][2048]
    const float* __restrict__ W2,     // [L][3][2048]
    const float* __restrict__ bf2,    // [L][3]
    const float* __restrict__ g2,     // [L][3]
    const float* __restrict__ beta2,  // [L][3]
    const float* __restrict__ Wfin,   // [3][9]
    float* __restrict__ out)          // [B][9]
{
    __shared__ float hbuf[TOK * 4];    // token hidden state, stride 4
    __shared__ float qkvbuf[TOK * 12]; // q(0..2) k(3..5) v(6..8); reused for z
    __shared__ float oatt[TOK * 4];    // attention output per token
    __shared__ float ffo[TOK * 4];     // FFN output per token

    const int lane = threadIdx.x;          // wave32
    const int batchBase = blockIdx.x * BT; // 512 blocks

    // ---- load x into per-wave token state ----
    for (int t = lane; t < TOK; t += 32) {
        int bi = t / S, s = t - bi * S;
        const float* xp = x + (size_t)(batchBase + bi) * (S * D) + s * D;
        hbuf[t * 4 + 0] = xp[0];
        hbuf[t * 4 + 1] = xp[1];
        hbuf[t * 4 + 2] = xp[2];
    }
    __syncthreads();

    for (int layer = 0; layer < NLAYER; ++layer) {
        const float* wqkv = Wqkv + layer * 9 * 3;
        const float* bq   = bqkv + layer * 9;

        // ---- QKV projection (tiny: 9x3 per token, VALU) ----
        for (int t = lane; t < TOK; t += 32) {
            float h0 = hbuf[t * 4 + 0], h1 = hbuf[t * 4 + 1], h2 = hbuf[t * 4 + 2];
#pragma unroll
            for (int e = 0; e < 9; ++e) {
                qkvbuf[t * 12 + e] = bq[e] + h0 * wqkv[e * 3 + 0]
                                           + h1 * wqkv[e * 3 + 1]
                                           + h2 * wqkv[e * 3 + 2];
            }
        }
        __syncthreads();

        // ---- attention, head_dim = 1 so scores are scalar products ----
        for (int u = lane; u < BT * H * S; u += 32) {
            int bi  = u / (H * S);
            int rem = u - bi * (H * S);
            int hh  = rem / S;
            int qi  = rem - hh * S;
            int base = bi * S;
            float q = qkvbuf[(base + qi) * 12 + hh];   // scale = 1/sqrt(1)
            float sc[9];
            float mx = -1e30f;
#pragma unroll
            for (int ki = 0; ki < 9; ++ki) {
                float sv = q * qkvbuf[(base + ki) * 12 + 3 + hh];
                sc[ki] = sv;
                mx = fmaxf(mx, sv);
            }
            float ssum = 0.f, ov = 0.f;
#pragma unroll
            for (int ki = 0; ki < 9; ++ki) {
                float ev = expf(sc[ki] - mx);
                ssum += ev;
                ov   += ev * qkvbuf[(base + ki) * 12 + 6 + hh];
            }
            oatt[(base + qi) * 4 + hh] = ov / ssum;
        }
        __syncthreads();

        // ---- out-projection + residual + LayerNorm1 ----
        {
            const float* wo  = Wo + layer * 9;
            const float* bop = bo + layer * 3;
            const float* gp  = g1 + layer * 3;
            const float* bp  = beta1 + layer * 3;
            for (int t = lane; t < TOK; t += 32) {
                float o0 = oatt[t * 4 + 0], o1 = oatt[t * 4 + 1], o2 = oatt[t * 4 + 2];
                float p0 = bop[0] + o0 * wo[0] + o1 * wo[1] + o2 * wo[2];
                float p1 = bop[1] + o0 * wo[3] + o1 * wo[4] + o2 * wo[5];
                float p2 = bop[2] + o0 * wo[6] + o1 * wo[7] + o2 * wo[8];
                float x0 = hbuf[t * 4 + 0] + p0;
                float x1 = hbuf[t * 4 + 1] + p1;
                float x2 = hbuf[t * 4 + 2] + p2;
                float m  = (x0 + x1 + x2) * (1.f / 3.f);
                float d0 = x0 - m, d1 = x1 - m, d2 = x2 - m;
                float vv = (d0 * d0 + d1 * d1 + d2 * d2) * (1.f / 3.f);
                float rs = rsqrtf(vv + LN_EPS);
                hbuf[t * 4 + 0] = d0 * rs * gp[0] + bp[0];
                hbuf[t * 4 + 1] = d1 * rs * gp[1] + bp[1];
                hbuf[t * 4 + 2] = d2 * rs * gp[2] + bp[2];
            }
        }
        __syncthreads();

        // ---- FFN: matmul1 via v_wmma_f32_16x16x32_f16, matmul2 via VALU+shfl ----
        {
            const float* w1L  = W1  + (size_t)layer * DFF * 3;
            const float* bf1L = bf1 + (size_t)layer * DFF;
            const float* w2L  = W2  + (size_t)layer * 3 * DFF;
            const int fl = lane & 15;

            for (int mt = 0; mt < MTILES; ++mt) {
                // A (16x32 f16): lanes 0-15 = rows M, VGPR halves 0..7 = K 0..7.
                // Only K=0..2 nonzero; lanes 16-31 carry K=8..15/24..31 -> all zero.
                v16h a;
#pragma unroll
                for (int i = 0; i < 16; ++i) a[i] = (_Float16)0.0f;
                if (lane < 16) {
                    int tok = mt * 16 + lane;
                    a[0] = (_Float16)hbuf[tok * 4 + 0];
                    a[1] = (_Float16)hbuf[tok * 4 + 1];
                    a[2] = (_Float16)hbuf[tok * 4 + 2];
                }

                float acc[8][3];
#pragma unroll
                for (int r = 0; r < 8; ++r) { acc[r][0] = 0.f; acc[r][1] = 0.f; acc[r][2] = 0.f; }

                for (int nf = 0; nf < NCHUNK; ++nf) {
                    int f = nf * 16 + fl;
                    // B (32x16 f16): lane = column N (f index), halves 0..7 = K 0..7, 8..15 = K 16..23.
                    v16h bm;
#pragma unroll
                    for (int i = 0; i < 16; ++i) bm[i] = (_Float16)0.0f;
                    if (lane < 16) {
                        const float* w1p = w1L + (size_t)f * 3;
                        bm[0] = (_Float16)w1p[0];
                        bm[1] = (_Float16)w1p[1];
                        bm[2] = (_Float16)w1p[2];
                    }
                    v8f c = {};
                    c = __builtin_amdgcn_wmma_f32_16x16x32_f16(
                            false, a, false, bm, (short)0, c, false, false);

                    // D layout: lane l<16 -> (M=r, N=l); l>=16 -> (M=r+8, N=l-16)
                    float bias = bf1L[f];
                    float w20 = w2L[f], w21 = w2L[DFF + f], w22 = w2L[2 * DFF + f];
#pragma unroll
                    for (int r = 0; r < 8; ++r) {
                        float v_ = fmaxf(c[r] + bias, 0.f);   // + bf1, relu
                        acc[r][0] += v_ * w20;
                        acc[r][1] += v_ * w21;
                        acc[r][2] += v_ * w22;
                    }
                }
                // reduce f across the 16 lanes of each half-wave
#pragma unroll
                for (int off = 1; off < 16; off <<= 1) {
#pragma unroll
                    for (int r = 0; r < 8; ++r) {
                        acc[r][0] += __shfl_xor(acc[r][0], off, 32);
                        acc[r][1] += __shfl_xor(acc[r][1], off, 32);
                        acc[r][2] += __shfl_xor(acc[r][2], off, 32);
                    }
                }
                int lh = lane & 15;
                if (lh < 8) {
                    int m = lh + ((lane >> 4) << 3);  // lower half: M=0..7, upper: M=8..15
                    int t = mt * 16 + m;
                    ffo[t * 4 + 0] = acc[lh][0];
                    ffo[t * 4 + 1] = acc[lh][1];
                    ffo[t * 4 + 2] = acc[lh][2];
                }
            }
        }
        __syncthreads();

        // ---- + bf2, residual, LayerNorm2 ----
        {
            const float* bfp = bf2 + layer * 3;
            const float* gp  = g2 + layer * 3;
            const float* bp  = beta2 + layer * 3;
            for (int t = lane; t < TOK; t += 32) {
                float x0 = hbuf[t * 4 + 0] + ffo[t * 4 + 0] + bfp[0];
                float x1 = hbuf[t * 4 + 1] + ffo[t * 4 + 1] + bfp[1];
                float x2 = hbuf[t * 4 + 2] + ffo[t * 4 + 2] + bfp[2];
                float m  = (x0 + x1 + x2) * (1.f / 3.f);
                float d0 = x0 - m, d1 = x1 - m, d2 = x2 - m;
                float vv = (d0 * d0 + d1 * d1 + d2 * d2) * (1.f / 3.f);
                float rs = rsqrtf(vv + LN_EPS);
                hbuf[t * 4 + 0] = d0 * rs * gp[0] + bp[0];
                hbuf[t * 4 + 1] = d1 * rs * gp[1] + bp[1];
                hbuf[t * 4 + 2] = d2 * rs * gp[2] + bp[2];
            }
        }
        __syncthreads();
    }

    // ---- policy head: per-token 3->9 logits + softmax (store z into qkvbuf) ----
    for (int t = lane; t < TOK; t += 32) {
        float h0 = hbuf[t * 4 + 0], h1 = hbuf[t * 4 + 1], h2 = hbuf[t * 4 + 2];
        float lg[9];
        float mx = -1e30f;
#pragma unroll
        for (int n = 0; n < 9; ++n) {
            lg[n] = h0 * Wfin[n] + h1 * Wfin[9 + n] + h2 * Wfin[18 + n];
            mx = fmaxf(mx, lg[n]);
        }
        float ssum = 0.f;
#pragma unroll
        for (int n = 0; n < 9; ++n) { lg[n] = expf(lg[n] - mx); ssum += lg[n]; }
        float inv = 1.f / ssum;
#pragma unroll
        for (int n = 0; n < 9; ++n) qkvbuf[t * 12 + n] = lg[n] * inv;
    }
    __syncthreads();

    // ---- product over the 9 tokens, final softmax, write out ----
    if (lane < BT) {
        int bi = lane;
        float u[9];
#pragma unroll
        for (int n = 0; n < 9; ++n) u[n] = 1.f;
        for (int s = 0; s < S; ++s) {
            int t = bi * S + s;
#pragma unroll
            for (int n = 0; n < 9; ++n) u[n] *= qkvbuf[t * 12 + n];
        }
        float mx = -1e30f;
#pragma unroll
        for (int n = 0; n < 9; ++n) mx = fmaxf(mx, u[n]);
        float ssum = 0.f;
#pragma unroll
        for (int n = 0; n < 9; ++n) { u[n] = expf(u[n] - mx); ssum += u[n]; }
        float inv = 1.f / ssum;
        float* op = out + (size_t)(batchBase + bi) * 9;
#pragma unroll
        for (int n = 0; n < 9; ++n) op[n] = u[n] * inv;
    }
}

extern "C" void kernel_launch(void* const* d_in, const int* in_sizes, int n_in,
                              void* d_out, int out_size, void* d_ws, size_t ws_size,
                              hipStream_t stream) {
    (void)in_sizes; (void)n_in; (void)out_size; (void)d_ws; (void)ws_size;
    const float* x     = (const float*)d_in[0];
    const float* Wqkv  = (const float*)d_in[1];
    const float* bqkv  = (const float*)d_in[2];
    const float* Wo    = (const float*)d_in[3];
    const float* bo    = (const float*)d_in[4];
    const float* g1    = (const float*)d_in[5];
    const float* beta1 = (const float*)d_in[6];
    const float* W1    = (const float*)d_in[7];
    const float* bf1   = (const float*)d_in[8];
    const float* W2    = (const float*)d_in[9];
    const float* bf2   = (const float*)d_in[10];
    const float* g2    = (const float*)d_in[11];
    const float* beta2 = (const float*)d_in[12];
    const float* Wfin  = (const float*)d_in[13];
    float* outp = (float*)d_out;

    dim3 grid(BATCH / BT);   // 512 single-wave workgroups
    dim3 block(32);
    hipLaunchKernelGGL(PolicyGradient_kernel, grid, block, 0, stream,
                       x, Wqkv, bqkv, Wo, bo, g1, beta1, W1, bf1, W2, bf2,
                       g2, beta2, Wfin, outp);
}
